// MultiHeadAttention_87187836109095
// MI455X (gfx1250) — matrix operand order repaired and checked by reading, not compiled, not run on hardware
//
#include <hip/hip_runtime.h>
#include <math.h>

typedef _Float16 v16h __attribute__((ext_vector_type(16)));
typedef _Float16 v8h  __attribute__((ext_vector_type(8)));
typedef _Float16 v4h  __attribute__((ext_vector_type(4)));
typedef float    v8f  __attribute__((ext_vector_type(8)));
typedef unsigned u32x4 __attribute__((ext_vector_type(4)));
typedef unsigned u32x8 __attribute__((ext_vector_type(8)));

#define WMMA_F32_F16(a, b, c) \
  __builtin_amdgcn_wmma_f32_16x16x32_f16(false, (a), false, (b), (short)0, (c), false, false)

static __device__ __forceinline__ v16h frag16(const _Float16* p, int lo, int hi) {
  v8h l = *(const v8h*)(p + lo);
  v8h h = *(const v8h*)(p + hi);
  return __builtin_shufflevector(l, h, 0,1,2,3,4,5,6,7,8,9,10,11,12,13,14,15);
}

// async 16B copy global -> LDS (ASYNCcnt).
static __device__ __forceinline__ void async_ld_b128(unsigned ldsoff, unsigned voff,
                                                     unsigned long long base) {
  asm volatile("global_load_async_to_lds_b128 %0, %1, %2"
               :: "v"(ldsoff), "v"(voff), "s"(base) : "memory");
}
static __device__ __forceinline__ void wait_async0() {
  asm volatile("s_wait_asynccnt 0x0" ::: "memory");
}
static __device__ __forceinline__ unsigned lds_off(const void* p) {
  return (unsigned)(unsigned long long)(uintptr_t)p;   // low 32 bits = LDS offset
}

// ---------------------------------------------------------------------------
// Tensor Data Mover: 2D f16 tile load, global (row-major, `stride` elements
// per row) -> LDS with hardware row padding. D# per CDNA5 ISA.
// g1w0 packs {data_size=2B, pad_enable, pad_interval, pad_amount}. Uniform
// operands only; issued once per wave; tracked by TENSORcnt.
// ---------------------------------------------------------------------------
static __device__ __forceinline__ void tdm_load_2d(
    unsigned ldsaddr, unsigned long long gaddr, unsigned g1w0,
    unsigned td0, unsigned td1, unsigned tile0, unsigned tile1, unsigned stride) {
  u32x4 g0;
  g0[0] = 1u;                                              // count=1, user mode
  g0[1] = ldsaddr;                                         // lds_addr
  g0[2] = (unsigned)gaddr;                                 // global_addr[31:0]
  g0[3] = ((unsigned)(gaddr >> 32) & 0x01FFFFFFu) | 0x80000000u; // [56:32]|type=2
  u32x8 g1;
  g1[0] = g1w0;                                            // mask=0|dsize|pad cfg
  g1[1] = (td0 & 0xFFFFu) << 16;                           // tensor_dim0[15:0]
  g1[2] = (td0 >> 16) | ((td1 & 0xFFFFu) << 16);           // td0 hi | td1 lo
  g1[3] = (td1 >> 16) | (tile0 << 16);                     // td1 hi | tile_dim0
  g1[4] = tile1 & 0xFFFFu;                                 // tile_dim1, tile_dim2=0
  g1[5] = stride;                                          // dim0_stride[31:0]
  g1[6] = 0;                                               // stride hi, dim1_stride
  g1[7] = 0;
  asm volatile("tensor_load_to_lds %0, %1" :: "s"(g0), "s"(g1) : "memory");
}
static __device__ __forceinline__ void wait_tensor0() {
#if __has_builtin(__builtin_amdgcn_s_wait_tensorcnt)
  __builtin_amdgcn_s_wait_tensorcnt(0);
#else
  asm volatile("s_wait_tensorcnt 0x0" ::: "memory");
#endif
}

// ---------------------------------------------------------------------------
// One-time converts: f32 -> f16 elementwise, and f32 [K,N] -> f16 [N,K].
// ---------------------------------------------------------------------------
__global__ void __launch_bounds__(256)
convert_f32_f16(const float* __restrict__ in, _Float16* __restrict__ out, int n4) {
  int i = blockIdx.x * 256 + threadIdx.x;
  if (i < n4) {
    float4 f = ((const float4*)in)[i];
    v4h h; h[0] = (_Float16)f.x; h[1] = (_Float16)f.y;
    h[2] = (_Float16)f.z; h[3] = (_Float16)f.w;
    ((v4h*)out)[i] = h;
  }
}

__global__ void __launch_bounds__(256)
transpose_convert(const float* __restrict__ W, _Float16* __restrict__ Wt) {
  constexpr int N = 1024;
  __shared__ float T[64][65];
  const int bn = blockIdx.x * 64, bk = blockIdx.y * 64;
  const int tid = threadIdx.x;
#pragma unroll
  for (int i = 0; i < 16; ++i) {                  // read W[k][n], coalesced in n
    int idx = tid + i * 256, r = idx >> 6, c = idx & 63;
    T[r][c] = W[(size_t)(bk + r) * N + bn + c];
  }
  __syncthreads();
#pragma unroll
  for (int i = 0; i < 16; ++i) {                  // write Wt[n][k], coalesced in k
    int idx = tid + i * 256, r = idx >> 6, c = idx & 63;
    Wt[(size_t)(bn + r) * N + bk + c] = (_Float16)T[c][r];
  }
}

// ---------------------------------------------------------------------------
// f16 GEMM: C[M,N] = A[M,K] @ Bw[N,K]^T (Bw rows are B-fragment rows).
// 128x128 block tile, 4 waves each 64x64, BK=32. Tiles staged by the Tensor
// Data Mover (wave 0 issues 2 descriptors/step), double-buffered LDS.
// LDS row stride 48 halves == 64B data + 32B TDM pad (interval=16DW, pad=8DW).
// ---------------------------------------------------------------------------
template <typename OT>
__global__ void __launch_bounds__(128)
gemm_f16(const _Float16* __restrict__ A, const _Float16* __restrict__ Bw,
         OT* __restrict__ C, int M, int N, int K) {
  constexpr int BM = 128, BN = 128, BK = 32, LDT = 48;
  constexpr unsigned G1W0 = (1u << 16)    // data_size = 2 bytes
                          | (1u << 20)    // pad_enable
                          | (3u << 22)    // pad_interval: 16 DWORDs (one row)
                          | (7u << 25);   // pad_amount:   8 DWORDs (32 B)
  __shared__ _Float16 As[2][BM * LDT];
  __shared__ _Float16 Bs[2][BN * LDT];

  const int tid  = threadIdx.x;
  const int w    = tid >> 5, lane = tid & 31;
  const int lrow = lane & 15, half = lane >> 4;
  const int wr   = w >> 1, wc = w & 1;                   // 2x2 waves of 64x64
  const int rowBase = blockIdx.y * BM;
  const int colBase = blockIdx.x * BN;

  const unsigned long long baseA =
      (unsigned long long)(uintptr_t)A + (unsigned long long)rowBase * K * 2;
  const unsigned long long baseB =
      (unsigned long long)(uintptr_t)Bw + (unsigned long long)colBase * K * 2;

  auto stage = [&](int buf, int kb) {
    if (w == 0) {
      tdm_load_2d(lds_off(&As[buf][0]), baseA + (unsigned long long)kb * 2,
                  G1W0, (unsigned)K, (unsigned)(M - rowBase), BK, BM, (unsigned)K);
      tdm_load_2d(lds_off(&Bs[buf][0]), baseB + (unsigned long long)kb * 2,
                  G1W0, (unsigned)K, (unsigned)(N - colBase), BK, BN, (unsigned)K);
    }
  };

  v8f acc[4][4] = {};
  stage(0, 0);
  int buf = 0;
  for (int kb = 0; kb < K; kb += BK) {
    if (w == 0) wait_tensor0();                          // tile[buf] landed
    __syncthreads();                                     // publish to all waves
    if (kb + BK < K) stage(buf ^ 1, kb + BK);            // DMA next tile

    v16h a[4];
#pragma unroll
    for (int mt = 0; mt < 4; ++mt)
      a[mt] = frag16(&As[buf][(wr * 64 + mt * 16 + lrow) * LDT],
                     half * 8, 16 + half * 8);
#pragma unroll
    for (int nt = 0; nt < 4; ++nt) {
      v16h b = frag16(&Bs[buf][(wc * 64 + nt * 16 + lrow) * LDT],
                      half * 8, 16 + half * 8);
#pragma unroll
      for (int mt = 0; mt < 4; ++mt)
        acc[mt][nt] = WMMA_F32_F16(a[mt], b, acc[mt][nt]);
    }
    buf ^= 1;
  }

#pragma unroll
  for (int mt = 0; mt < 4; ++mt)
#pragma unroll
    for (int nt = 0; nt < 4; ++nt)
#pragma unroll
      for (int r = 0; r < 8; ++r) {
        int row = rowBase + wr * 64 + mt * 16 + r + 8 * half;
        int col = colBase + wc * 64 + nt * 16 + lrow;
        C[(size_t)row * N + col] = (OT)acc[mt][nt][r];
      }
}

// ---------------------------------------------------------------------------
// Causal flash attention. Q/K/V f16 [B,S,H*HD]. Block = 64 q-rows x head x
// batch; 4 waves x 16 q-rows. K tile staged with async-to-LDS; V transposed
// through VGPRs; online softmax in f32; all matmuls WMMA f16.
// ---------------------------------------------------------------------------
__global__ void __launch_bounds__(128)
flash_attn(const _Float16* __restrict__ Q, const _Float16* __restrict__ Km,
           const _Float16* __restrict__ V, _Float16* __restrict__ O) {
  constexpr int S = 2048, D = 1024, HD = 64, LDK = 72;   // 144B LDS rows
  __shared__ _Float16 Ks[64 * LDK];
  __shared__ _Float16 Vt[64 * LDK];
  __shared__ _Float16 Ps[4 * 16 * LDK];

  const int qblk = blockIdx.x, h = blockIdx.y, b = blockIdx.z;
  const int tid  = threadIdx.x;
  const int w    = tid >> 5, lane = tid & 31;
  const int lrow = lane & 15, half = lane >> 4;

  const size_t bhoff = (size_t)b * S * D + h * HD;
  const int qw = qblk * 64 + w * 16;
  const unsigned long long baseK =
      (unsigned long long)(uintptr_t)Km + (unsigned long long)bhoff * 2;

  const _Float16* qp = Q + bhoff + (size_t)(qw + lrow) * D;
  const v16h aQ0 = frag16(qp, half * 8, 16 + half * 8);
  const v16h aQ1 = frag16(qp, 32 + half * 8, 48 + half * 8);

  float m_run[8], l_run[8];
  v8f o[4] = {};
#pragma unroll
  for (int r = 0; r < 8; ++r) { m_run[r] = -1e30f; l_run[r] = 0.f; }

  for (int kt = 0; kt <= qblk; ++kt) {
    const int kbase = kt * 64;
#pragma unroll
    for (int i = 0; i < 4; ++i) {                        // async K tile
      int idx = tid + i * 128, r = idx >> 3, c8 = idx & 7;
      async_ld_b128(lds_off(&Ks[r * LDK + c8 * 8]),
                    (unsigned)(((kbase + r) * D + c8 * 8) * 2), baseK);
    }
#pragma unroll
    for (int i = 0; i < 4; ++i) {                        // V^T via VGPRs
      int idx = tid + i * 128, r = idx >> 3, c8 = idx & 7;
      v8h vv = *(const v8h*)(V + bhoff + (size_t)(kbase + r) * D + c8 * 8);
#pragma unroll
      for (int j = 0; j < 8; ++j) Vt[(c8 * 8 + j) * LDK + r] = vv[j];
    }
    if (kt < qblk) {                                     // prefetch next tiles
      const size_t nx = bhoff + (size_t)(kbase + 64 + (tid & 63)) * D;
      __builtin_prefetch(Km + nx, 0, 0);
      __builtin_prefetch(V + nx, 0, 0);
    }
    wait_async0();
    __syncthreads();

    // S = Q @ K^T
    v8f s[4] = {};
#pragma unroll
    for (int nt = 0; nt < 4; ++nt) {
      const _Float16* kp = &Ks[(nt * 16 + lrow) * LDK];
      v16h b0 = frag16(kp, half * 8, 16 + half * 8);
      v16h b1 = frag16(kp, 32 + half * 8, 48 + half * 8);
      s[nt] = WMMA_F32_F16(aQ0, b0, s[nt]);
      s[nt] = WMMA_F32_F16(aQ1, b1, s[nt]);
    }

    // scale, causal mask, online softmax (row = r + 8*half, col = lrow)
#pragma unroll
    for (int r = 0; r < 8; ++r) {
      const int qrow = qw + r + 8 * half;
      float rm = -1e30f;
#pragma unroll
      for (int nt = 0; nt < 4; ++nt) {
        const int key = kbase + nt * 16 + lrow;
        float v = s[nt][r] * 0.125f;                     // 1/sqrt(HD)
        v = (key > qrow) ? -1e30f : v;
        s[nt][r] = v;
        rm = fmaxf(rm, v);
      }
#pragma unroll
      for (int msk = 1; msk < 16; msk <<= 1) rm = fmaxf(rm, __shfl_xor(rm, msk, 32));
      const float mn = fmaxf(m_run[r], rm);
      const float corr = __expf(m_run[r] - mn);
      float rs = 0.f;
#pragma unroll
      for (int nt = 0; nt < 4; ++nt) {
        float p = __expf(s[nt][r] - mn);
        s[nt][r] = p;
        rs += p;
      }
#pragma unroll
      for (int msk = 1; msk < 16; msk <<= 1) rs += __shfl_xor(rs, msk, 32);
      l_run[r] = l_run[r] * corr + rs;
      m_run[r] = mn;
#pragma unroll
      for (int nt = 0; nt < 4; ++nt) o[nt][r] *= corr;
    }

    // re-layout P: C-fragment -> A-fragment via LDS
#pragma unroll
    for (int r = 0; r < 8; ++r)
#pragma unroll
      for (int nt = 0; nt < 4; ++nt)
        Ps[(w * 16 + r + 8 * half) * LDK + nt * 16 + lrow] = (_Float16)s[nt][r];
    __syncthreads();

    const _Float16* pp = &Ps[(w * 16 + lrow) * LDK];
    v16h aP0 = frag16(pp, half * 8, 16 + half * 8);
    v16h aP1 = frag16(pp, 32 + half * 8, 48 + half * 8);
#pragma unroll
    for (int nt = 0; nt < 4; ++nt) {                     // O += P @ V
      const _Float16* vp = &Vt[(nt * 16 + lrow) * LDK];
      v16h b0 = frag16(vp, half * 8, 16 + half * 8);
      v16h b1 = frag16(vp, 32 + half * 8, 48 + half * 8);
      o[nt] = WMMA_F32_F16(aP0, b0, o[nt]);
      o[nt] = WMMA_F32_F16(aP1, b1, o[nt]);
    }
    __syncthreads();
  }

#pragma unroll
  for (int r = 0; r < 8; ++r) {
    const float inv = 1.f / l_run[r];
    const int qrow = qw + r + 8 * half;
#pragma unroll
    for (int nt = 0; nt < 4; ++nt)
      O[bhoff + (size_t)qrow * D + nt * 16 + lrow] = (_Float16)(o[nt][r] * inv);
  }
}

// ---------------------------------------------------------------------------
extern "C" void kernel_launch(void* const* d_in, const int* in_sizes, int n_in,
                              void* d_out, int out_size, void* d_ws, size_t ws_size,
                              hipStream_t stream) {
  const float* x  = (const float*)d_in[0];
  const float* Wq = (const float*)d_in[1];
  const float* Wk = (const float*)d_in[2];
  const float* Wv = (const float*)d_in[3];
  const float* Wo = (const float*)d_in[4];

  constexpr int Bn = 4, S = 2048, D = 1024;
  constexpr size_t MD = (size_t)Bn * S * D;              // 8388608
  constexpr size_t WD = (size_t)D * D;                   // 1048576

  _Float16* Xw  = (_Float16*)d_ws;       // f16 x            (16 MB)
  _Float16* Wqt = Xw  + MD;              // f16 W_q^T        ( 2 MB)
  _Float16* Wkt = Wqt + WD;
  _Float16* Wvt = Wkt + WD;
  _Float16* Wot = Wvt + WD;              // f16 W_o (rows)   ( 2 MB)
  _Float16* Qw  = Wot + WD;              // f16 Q/K/V        (16 MB each)
  _Float16* Kw  = Qw  + MD;
  _Float16* Vw  = Kw  + MD;
  _Float16* Aw  = Vw  + MD;              // f16 attn out     (16 MB)

  convert_f32_f16<<<(int)(MD / 4 / 256), 256, 0, stream>>>(x, Xw, (int)(MD / 4));
  dim3 gt(16, 16);
  transpose_convert<<<gt, 256, 0, stream>>>(Wq, Wqt);
  transpose_convert<<<gt, 256, 0, stream>>>(Wk, Wkt);
  transpose_convert<<<gt, 256, 0, stream>>>(Wv, Wvt);
  convert_f32_f16<<<(int)(WD / 4 / 256), 256, 0, stream>>>(Wo, Wot, (int)(WD / 4));

  dim3 gg(D / 128, (Bn * S) / 128), bb(128);
  gemm_f16<_Float16><<<gg, bb, 0, stream>>>(Xw, Wqt, Qw, Bn * S, D, D);
  gemm_f16<_Float16><<<gg, bb, 0, stream>>>(Xw, Wkt, Kw, Bn * S, D, D);
  gemm_f16<_Float16><<<gg, bb, 0, stream>>>(Xw, Wvt, Vw, Bn * S, D, D);

  dim3 ga(S / 64, 16, Bn), ba(128);
  flash_attn<<<ga, ba, 0, stream>>>(Qw, Kw, Vw, Aw);

  gemm_f16<float><<<gg, bb, 0, stream>>>(Aw, Wot, (float*)d_out, Bn * S, D, D);
}
